// HATBlock_41738492182644
// MI455X (gfx1250) — matrix operand order repaired and checked by reading, not compiled, not run on hardware
//
#include <hip/hip_runtime.h>

// ---------------------------------------------------------------------------
// HAT block for MI455X (gfx1250).  All GEMMs + both attention einsums on
// v_wmma_f32_16x16x32_f16, with vectorized fragment loads:
//   * A operands: two contiguous v8h (b128) LDS loads per lane
//   * B operands (weights): pre-packed fragment-order in workspace -> one
//     contiguous 32B global load per lane, fully coalesced
//   * attention K/V staged in head-padded LDS (K:[64|144][32], V^T:[32][keys])
//     so no per-element masking in the hot loops
// ---------------------------------------------------------------------------

typedef __attribute__((ext_vector_type(16))) _Float16 v16h;
typedef __attribute__((ext_vector_type(8)))  _Float16 v8h;
typedef __attribute__((ext_vector_type(8)))  float    v8f;

#define CH   192
#define NH   8
#define HD   24
#define WS   8
#define NTOK 64
#define OV   2
#define WSO  12
#define NKEY 144
#define IMH  256
#define IMW  256
#define NWIN 4096
#define ATT_SCALE 0.2041241452319315f  // 24^-0.5

__device__ __forceinline__ int xidx(int b, int c, int h, int w) {
  return ((b * CH + c) * IMH + h) * IMW + w;
}
__device__ __forceinline__ int refl(int i, int n) {
  i = (i < 0) ? -i : i;
  if (i >= n) i = 2 * n - 2 - i;
  return i;
}
__device__ __forceinline__ v16h vcat(v8h lo, v8h hi) {
  v16h a;
#pragma unroll
  for (int e = 0; e < 8; ++e) { a[e] = lo[e]; a[e + 8] = hi[e]; }
  return a;
}

// A fragment (16x32 f16): lane = half*16+m ; element e -> K = half*16 + e
// => per lane one contiguous 16-halfword run at col0+half*16.
__device__ __forceinline__ v16h load_a_fast(const _Float16* __restrict__ base,
                                            int ld, int row0, int col0) {
  const int lane = threadIdx.x & 31;
  const int m = lane & 15, half = lane >> 4;
  const _Float16* rp = base + (row0 + m) * ld + col0 + half * 16;
  return vcat(*(const v8h*)rp, *(const v8h*)(rp + 8));
}
// B fragment where contraction dim is contiguous per lane:
// B[k][n] = base[n*ld + k0 + k]  (k = half*16 + e)
__device__ __forceinline__ v16h load_bk_fast(const _Float16* __restrict__ base,
                                             int ld, int k0) {
  const int lane = threadIdx.x & 31;
  const int n = lane & 15, half = lane >> 4;
  const _Float16* rp = base + n * ld + k0 + half * 16;
  return vcat(*(const v8h*)rp, *(const v8h*)(rp + 8));
}
// Pre-packed weight fragment: tile (kt,nt) stored as 512 contiguous f16,
// lane-major: value(lane,e) at tile_base + lane*16 + e.
__device__ __forceinline__ v16h load_b_packed(const _Float16* __restrict__ base,
                                              int ktiles, int kt, int nt) {
  const _Float16* p = base + (((size_t)nt * ktiles + kt) << 9)
                           + ((threadIdx.x & 31) << 4);
  return vcat(*(const v8h*)p, *(const v8h*)(p + 8));
}
// C/D (16x16 f32): lane = half*16+n ; reg r -> row r+8*half
__device__ __forceinline__ void store_d_f32(float* base, int ld, int row0, int col0, v8f d) {
  const int lane = threadIdx.x & 31;
  const int n = lane & 15, half = lane >> 4;
#pragma unroll
  for (int r = 0; r < 8; ++r)
    base[(row0 + r + 8 * half) * ld + col0 + n] = d[r];
}
__device__ __forceinline__ void store_d_f16(_Float16* base, int ld, int row0, int col0,
                                            int colmax, v8f d) {
  const int lane = threadIdx.x & 31;
  const int n = lane & 15, half = lane >> 4;
  const int c = col0 + n;
  if (c >= colmax) return;
#pragma unroll
  for (int r = 0; r < 8; ++r)
    base[(row0 + r + 8 * half) * ld + c] = (_Float16)d[r];
}
__device__ __forceinline__ v8f wmma32(v16h a, v16h b, v8f c) {
  return __builtin_amdgcn_wmma_f32_16x16x32_f16(false, a, false, b, (short)0, c,
                                                false, false);
}

// Scatter fused-QKV D tile into head-padded LDS: q32/k32 [h][64][32], vT [h][32][64]
__device__ __forceinline__ void store_qkv_tile(_Float16* q32, _Float16* k32,
                                               _Float16* vT, int mt, int nt, v8f d) {
  const int lane = threadIdx.x & 31;
  const int n = lane & 15, half = lane >> 4;
  const int ccc = nt * 16 + n;
  const int g = ccc / CH, rem = ccc % CH;
  const int h = rem / HD, dd = rem % HD;
#pragma unroll
  for (int r = 0; r < 8; ++r) {
    int t = mt * 16 + r + 8 * half;
    if (g == 0)      q32[h * (NTOK * 32) + t * 32 + dd]   = (_Float16)d[r];
    else if (g == 1) k32[h * (NTOK * 32) + t * 32 + dd]   = (_Float16)d[r];
    else             vT [h * (32 * NTOK) + dd * NTOK + t] = (_Float16)d[r];
  }
}

// ---------------- weight pack kernel: f32 row-major -> f16 fragment-order ---
__global__ void pack_b_kernel(const float* __restrict__ src, _Float16* __restrict__ dst,
                              int K, int N) {
  int i = blockIdx.x * 256 + threadIdx.x;
  if (i >= K * N) return;
  int tile = i >> 9, r = i & 511;
  int lane = r >> 4, e = r & 15;
  int half = lane >> 4, n = lane & 15;
  int ktiles = K >> 5;
  int nt = tile / ktiles, kt = tile % ktiles;
  dst[i] = (_Float16)src[(kt * 32 + half * 16 + e) * N + nt * 16 + n];
}

// ===========================================================================
// Kernel 1: fused windowed branch.  out = x + alpha * (WMSA+MLP result)
// LDS plan (285728 B):
//   [0,131072)        : x window f32 -> scores 8x64x64 f32 (P f16 in place) -> xw1
//   [131072,155648)   : a16/h16 (64x192)
//   [155648,188416)   : q32 [8][64][32]      (g16 64x384 reuses 155648.. in MLP)
//   [188416,221184)   : k32 [8][64][32]
//   [221184,253952)   : vT  [8][32][64]
//   [253952,278528)   : o16 (64x192)
//   [278528,285728)   : rpe (225x8 f32)
// ===========================================================================
#define K1_A    131072
#define K1_Q32  155648
#define K1_K32  188416
#define K1_VT   221184
#define K1_O    253952
#define K1_RPE  278528
#define K1_SMEM 285728

__global__ __launch_bounds__(256) void hat_win_kernel(
    const float* __restrict__ x,
    const float* __restrict__ n1g, const float* __restrict__ n1b,
    const float* __restrict__ n2g, const float* __restrict__ n2b,
    const _Float16* __restrict__ qkvw, const _Float16* __restrict__ projw,
    const float* __restrict__ rpe,
    const _Float16* __restrict__ w1, const _Float16* __restrict__ w2,
    const float* __restrict__ alpha_p, float* __restrict__ out)
{
  extern __shared__ char smem[];
  float*    s_x   = (float*)(smem);            // 64x192 f32 / scores / xw1
  float*    s_sc  = (float*)(smem);
  _Float16* s_a   = (_Float16*)(smem + K1_A);
  _Float16* s_q32 = (_Float16*)(smem + K1_Q32);
  _Float16* s_k32 = (_Float16*)(smem + K1_K32);
  _Float16* s_vT  = (_Float16*)(smem + K1_VT);
  _Float16* s_o   = (_Float16*)(smem + K1_O);
  float*    s_rpe = (float*)(smem + K1_RPE);

  const int win = blockIdx.x;
  const int b = win >> 10, wy = (win >> 5) & 31, wx = win & 31;
  const int h0 = wy * WS, w0 = wx * WS;
  const int tid = threadIdx.x, wave = tid >> 5;
  const float alpha = alpha_p[0];

  // ---- phase 0: stage window + rpe, zero padded q/k/v staging ----
  for (int i = tid; i < NTOK * CH; i += 256) {
    int c = i / NTOK, t = i % NTOK;
    s_x[t * CH + c] = x[xidx(b, c, h0 + (t >> 3), w0 + (t & 7))];
  }
  for (int i = tid; i < 225 * NH; i += 256) s_rpe[i] = rpe[i];
  for (int i = tid * 4; i < 3 * 32768; i += 1024)
    *(unsigned int*)(smem + K1_Q32 + i) = 0u;
  __syncthreads();

  // ---- phase 1: LayerNorm1 -> a16 ----
  if (tid < NTOK) {
    const float* r = s_x + tid * CH;
    float m = 0.f;
    for (int c = 0; c < CH; ++c) m += r[c];
    m *= (1.0f / CH);
    float v = 0.f;
    for (int c = 0; c < CH; ++c) { float d = r[c] - m; v += d * d; }
    float rs = rsqrtf(v * (1.0f / CH) + 1e-5f);
    for (int c = 0; c < CH; ++c)
      s_a[tid * CH + c] = (_Float16)((r[c] - m) * rs * n1g[c] + n1b[c]);
  }
  __syncthreads();

  // ---- phase 2: QKV GEMM (64x192)x(192x576), scatter into padded staging ----
  for (int tile = wave; tile < 4 * 36; tile += 8) {
    int mt = tile / 36, nt = tile % 36;
    v8f acc = {};
#pragma unroll
    for (int kt = 0; kt < 6; ++kt) {
      v16h a = load_a_fast(s_a, CH, mt * 16, kt * 32);
      v16h bm = load_b_packed(qkvw, 6, kt, nt);
      acc = wmma32(a, bm, acc);
    }
    store_qkv_tile(s_q32, s_k32, s_vT, mt, nt, acc);
  }
  __syncthreads();

  // ---- phase 3: S = Q K^T per head (wave == head), K padded 24->32 ----
  {
    const int h = wave;
    const _Float16* Qh = s_q32 + h * (NTOK * 32);
    const _Float16* Kh = s_k32 + h * (NTOK * 32);
    float* Sh = s_sc + h * 4096;
#pragma unroll
    for (int tile = 0; tile < 16; ++tile) {
      int mt = tile >> 2, nt = tile & 3;
      v16h a = load_a_fast(Qh, 32, mt * 16, 0);
      v16h bm = load_bk_fast(Kh + nt * 16 * 32, 32, 0);
      v8f acc = {};
      acc = wmma32(a, bm, acc);
      store_d_f32(Sh, 64, mt * 16, nt * 16, acc);
    }
  }
  __syncthreads();

  // ---- phase 4a: softmax (+RPE), P written f16 in place over score row ----
  for (int row = tid; row < NH * NTOK; row += 256) {
    int h = row >> 6, q = row & 63;
    int qy = q >> 3, qx = q & 7;
    float* Sr = s_sc + h * 4096 + q * 64;
    float mx = -1e30f;
    for (int k = 0; k < 64; ++k) {
      int bias = ((qy - (k >> 3)) + 7) * 15 + ((qx - (k & 7)) + 7);
      mx = fmaxf(mx, Sr[k] * ATT_SCALE + s_rpe[bias * NH + h]);
    }
    float sum = 0.f;
    for (int k = 0; k < 64; ++k) {
      int bias = ((qy - (k >> 3)) + 7) * 15 + ((qx - (k & 7)) + 7);
      sum += __expf(Sr[k] * ATT_SCALE + s_rpe[bias * NH + h] - mx);
    }
    float inv = 1.0f / sum;
    volatile const float* vSr = Sr;
    volatile _Float16* vPr = (_Float16*)Sr;   // row-exclusive in-place f16
    for (int k = 0; k < 64; ++k) {
      int bias = ((qy - (k >> 3)) + 7) * 15 + ((qx - (k & 7)) + 7);
      float sv = vSr[k];
      vPr[k] = (_Float16)(__expf(sv * ATT_SCALE + s_rpe[bias * NH + h] - mx) * inv);
    }
  }
  __syncthreads();

  // ---- phase 4b: O = P V per head (P ld=128 f16 view of score rows) ----
  {
    const int h = wave;
    const _Float16* Ph = (const _Float16*)(s_sc + h * 4096);
    const _Float16* VTh = s_vT + h * (32 * NTOK);
#pragma unroll
    for (int mt = 0; mt < 4; ++mt)
#pragma unroll
      for (int nt = 0; nt < 2; ++nt) {
        v8f acc = {};
#pragma unroll
        for (int kk = 0; kk < 64; kk += 32) {
          v16h a = load_a_fast(Ph, 128, mt * 16, kk);
          v16h bm = load_bk_fast(VTh, NTOK, kk);
          acc = wmma32(a, bm, acc);
        }
        store_d_f16(s_o, CH, mt * 16, h * HD + nt * 16, h * HD + HD, acc);
      }
  }
  __syncthreads();

  // ---- phase 5: proj -> xw1 (f32 at offset 0) ; add residual x ----
  for (int tile = wave; tile < 48; tile += 8) {
    int mt = tile / 12, nt = tile % 12;
    v8f acc = {};
#pragma unroll
    for (int kt = 0; kt < 6; ++kt) {
      v16h a = load_a_fast(s_o, CH, mt * 16, kt * 32);
      v16h bm = load_b_packed(projw, 6, kt, nt);
      acc = wmma32(a, bm, acc);
    }
    store_d_f32(s_x, CH, mt * 16, nt * 16, acc);
  }
  __syncthreads();
  for (int i = tid; i < NTOK * CH; i += 256) {
    int c = i / NTOK, t = i % NTOK;
    s_x[t * CH + c] += x[xidx(b, c, h0 + (t >> 3), w0 + (t & 7))];
  }
  __syncthreads();

  // ---- phase 6: LN2 -> h16 ; g = gelu(h @ w1) (64x384 f16, reuses q32 area) ----
  if (tid < NTOK) {
    const float* r = s_x + tid * CH;
    float m = 0.f;
    for (int c = 0; c < CH; ++c) m += r[c];
    m *= (1.0f / CH);
    float v = 0.f;
    for (int c = 0; c < CH; ++c) { float d = r[c] - m; v += d * d; }
    float rs = rsqrtf(v * (1.0f / CH) + 1e-5f);
    for (int c = 0; c < CH; ++c)
      s_a[tid * CH + c] = (_Float16)((r[c] - m) * rs * n2g[c] + n2b[c]);
  }
  __syncthreads();
  _Float16* s_g = (_Float16*)(smem + K1_Q32);
  for (int tile = wave; tile < 4 * 24; tile += 8) {
    int mt = tile / 24, nt = tile % 24;
    v8f acc = {};
#pragma unroll
    for (int kt = 0; kt < 6; ++kt) {
      v16h a = load_a_fast(s_a, CH, mt * 16, kt * 32);
      v16h bm = load_b_packed(w1, 6, kt, nt);
      acc = wmma32(a, bm, acc);
    }
#pragma unroll
    for (int r = 0; r < 8; ++r) {
      float v = acc[r];
      acc[r] = 0.5f * v * (1.0f + erff(v * 0.70710678f));  // exact GELU
    }
    store_d_f16(s_g, 2 * CH, mt * 16, nt * 16, 2 * CH, acc);
  }
  __syncthreads();

  // ---- phase 7: y = g @ w2 ; out = x + alpha*(xw1 + y) ----
  for (int tile = wave; tile < 48; tile += 8) {
    int mt = tile / 12, nt = tile % 12;
    v8f acc = {};
#pragma unroll
    for (int kt = 0; kt < 12; ++kt) {
      v16h a = load_a_fast(s_g, 2 * CH, mt * 16, kt * 32);
      v16h bm = load_b_packed(w2, 12, kt, nt);
      acc = wmma32(a, bm, acc);
    }
    const int lane = threadIdx.x & 31;
    const int n = lane & 15, half = lane >> 4;
    const int c = nt * 16 + n;
#pragma unroll
    for (int r = 0; r < 8; ++r) {
      int t = mt * 16 + r + 8 * half;
      int gi = xidx(b, c, h0 + (t >> 3), w0 + (t & 7));
      out[gi] = x[gi] + alpha * (s_x[t * CH + c] + acc[r]);
    }
  }
}

// ===========================================================================
// Kernel 2: OCA branch.  out += (1-alpha) * x_oca
// LDS plan (305152 B):
//   [0,81920)         : x window f32 (phase0) -> vT [8][32][160]
//   [81920,114688)    : q32 [8][64][32]
//   [114688,169984)   : xuf16 144x192 -> scores slot0 (64x144 f32, P in place)
//   [169984,243712)   : k32 [8][144][32]   (a16 64x192 overlays start, ph0-1)
//   [243712,280576)   : scores slot1
//   [280576,305152)   : o16 (64x192)
// ===========================================================================
#define K3_VT   0
#define K3_Q32  81920
#define K3_XUF  114688
#define K3_K32  169984
#define K3_SC1  243712
#define K3_O    280576
#define K3_SMEM 305152

__global__ __launch_bounds__(256) void hat_oca_kernel(
    const float* __restrict__ x,
    const float* __restrict__ ng, const float* __restrict__ nb,
    const _Float16* __restrict__ qw, const _Float16* __restrict__ kvw,
    const _Float16* __restrict__ projw,
    const float* __restrict__ alpha_p, float* __restrict__ out)
{
  extern __shared__ char smem[];
  float*    s_x   = (float*)(smem);
  _Float16* s_vT  = (_Float16*)(smem + K3_VT);   // [8][32][160]
  _Float16* s_q32 = (_Float16*)(smem + K3_Q32);  // [8][64][32]
  _Float16* s_xuf = (_Float16*)(smem + K3_XUF);  // 144x192
  float*    s_sc0 = (float*)(smem + K3_XUF);     // 64x144 f32
  _Float16* s_a   = (_Float16*)(smem + K3_K32);  // 64x192 (phases 0-1)
  _Float16* s_k32 = (_Float16*)(smem + K3_K32);  // [8][144][32]
  float*    s_sc1 = (float*)(smem + K3_SC1);
  _Float16* s_o   = (_Float16*)(smem + K3_O);

  const int win = blockIdx.x;
  const int b = win >> 10, wy = (win >> 5) & 31, wx = win & 31;
  const int h0 = wy * WS, w0 = wx * WS;
  const int tid = threadIdx.x, wave = tid >> 5;
  const float am1 = 1.0f - alpha_p[0];

  // ---- phase 0: stage window ; zero q32 ; LN_oca -> a16 ----
  for (int i = tid; i < NTOK * CH; i += 256) {
    int c = i / NTOK, t = i % NTOK;
    s_x[t * CH + c] = x[xidx(b, c, h0 + (t >> 3), w0 + (t & 7))];
  }
  for (int i = tid * 4; i < 32768; i += 1024)
    *(unsigned int*)(smem + K3_Q32 + i) = 0u;
  __syncthreads();
  if (tid < NTOK) {
    const float* r = s_x + tid * CH;
    float m = 0.f;
    for (int c = 0; c < CH; ++c) m += r[c];
    m *= (1.0f / CH);
    float v = 0.f;
    for (int c = 0; c < CH; ++c) { float d = r[c] - m; v += d * d; }
    float rs = rsqrtf(v * (1.0f / CH) + 1e-5f);
    for (int c = 0; c < CH; ++c)
      s_a[tid * CH + c] = (_Float16)((r[c] - m) * rs * ng[c] + nb[c]);
  }
  __syncthreads();

  // ---- phase 1: q = a @ q_w, scatter into padded q32 ----
  for (int tile = wave; tile < 48; tile += 8) {
    int mt = tile / 12, nt = tile % 12;
    v8f acc = {};
#pragma unroll
    for (int kt = 0; kt < 6; ++kt) {
      v16h a = load_a_fast(s_a, CH, mt * 16, kt * 32);
      v16h bm = load_b_packed(qw, 6, kt, nt);
      acc = wmma32(a, bm, acc);
    }
    {
      const int lane = threadIdx.x & 31;
      const int n = lane & 15, half = lane >> 4;
      const int ccc = nt * 16 + n;
      const int h = ccc / HD, dd = ccc % HD;
#pragma unroll
      for (int r = 0; r < 8; ++r) {
        int t = mt * 16 + r + 8 * half;
        s_q32[h * (NTOK * 32) + t * 32 + dd] = (_Float16)acc[r];
      }
    }
  }
  __syncthreads();

  // ---- phase 2: reflect-pad unfold -> xuf16 ; zero k32 + vT (s_x dead) ----
  for (int i = tid; i < NKEY * CH; i += 256) {
    int c = i / NKEY, s = i % NKEY;
    int gy = refl(h0 + (s / WSO) - OV, IMH);
    int gx = refl(w0 + (s % WSO) - OV, IMW);
    s_xuf[s * CH + c] = (_Float16)x[xidx(b, c, gy, gx)];
  }
  for (int i = tid * 4; i < 73728; i += 1024)
    *(unsigned int*)(smem + K3_K32 + i) = 0u;
  for (int i = tid * 4; i < 81920; i += 1024)
    *(unsigned int*)(smem + K3_VT + i) = 0u;
  __syncthreads();

  // ---- phase 3: kv = xuf @ kv_w (144x192)x(192x384), scatter padded ----
  for (int tile = wave; tile < 9 * 24; tile += 8) {
    int mt = tile / 24, nt = tile % 24;
    v8f acc = {};
#pragma unroll
    for (int kt = 0; kt < 6; ++kt) {
      v16h a = load_a_fast(s_xuf, CH, mt * 16, kt * 32);
      v16h bm = load_b_packed(kvw, 6, kt, nt);
      acc = wmma32(a, bm, acc);
    }
    {
      const int lane = threadIdx.x & 31;
      const int n = lane & 15, half = lane >> 4;
      const int ccc = nt * 16 + n;
      const int g = ccc / CH, rem = ccc % CH;
      const int h = rem / HD, dd = rem % HD;
#pragma unroll
      for (int r = 0; r < 8; ++r) {
        int t = mt * 16 + r + 8 * half;   // 0..143
        if (g == 0) s_k32[h * (NKEY * 32) + t * 32 + dd] = (_Float16)acc[r];
        else        s_vT [h * (32 * 160) + dd * 160 + t] = (_Float16)acc[r];
      }
    }
  }
  __syncthreads();

  // ---- phase 4: attention, 4 rounds x 2 heads (4 waves per head) ----
  for (int hr = 0; hr < 4; ++hr) {
    const int sub = wave >> 2, wi = wave & 3;
    const int h = hr * 2 + sub;
    float* Sh = (sub == 0) ? s_sc0 : s_sc1;
    const _Float16* Qh = s_q32 + h * (NTOK * 32);
    const _Float16* Kh = s_k32 + h * (NKEY * 32);
    const _Float16* VTh = s_vT + h * (32 * 160);

    // S = q_h k_h^T  (64x144), contraction 24 padded to 32 with zeros
#pragma unroll
    for (int ntk = 0; ntk < 9; ++ntk) {
      v16h a = load_a_fast(Qh, 32, wi * 16, 0);
      v16h bm = load_bk_fast(Kh + ntk * 16 * 32, 32, 0);
      v8f acc = {};
      acc = wmma32(a, bm, acc);
      store_d_f32(Sh, NKEY, wi * 16, ntk * 16, acc);
    }
    __syncthreads();

    // softmax rows, P f16 in place (ld 288), pad cols 144..159 zeroed
    if (tid < 128) {
      int s2 = tid >> 6, q = tid & 63;
      float* Sr = ((s2 == 0) ? s_sc0 : s_sc1) + q * NKEY;
      float mx = -1e30f;
      for (int k = 0; k < NKEY; ++k) mx = fmaxf(mx, Sr[k] * ATT_SCALE);
      float sum = 0.f;
      for (int k = 0; k < NKEY; ++k) sum += __expf(Sr[k] * ATT_SCALE - mx);
      float inv = 1.0f / sum;
      volatile const float* vSr = Sr;
      volatile _Float16* vPr = (_Float16*)Sr;
      for (int k = 0; k < NKEY; ++k) {
        float sv = vSr[k];
        vPr[k] = (_Float16)(__expf(sv * ATT_SCALE - mx) * inv);
      }
      for (int k = NKEY; k < 160; ++k) vPr[k] = (_Float16)0.0f;
    }
    __syncthreads();

    // O = P V   (K padded 144->160, both operands zero-padded)
    const _Float16* Ph = (const _Float16*)((sub == 0) ? s_sc0 : s_sc1);
#pragma unroll
    for (int nt = 0; nt < 2; ++nt) {
      v8f acc = {};
#pragma unroll
      for (int kk = 0; kk < 160; kk += 32) {
        v16h a = load_a_fast(Ph, 2 * NKEY, wi * 16, kk);
        v16h bm = load_bk_fast(VTh, 160, kk);
        acc = wmma32(a, bm, acc);
      }
      store_d_f16(s_o, CH, wi * 16, h * HD + nt * 16, h * HD + HD, acc);
    }
    __syncthreads();
  }

  // ---- phase 5: proj + residual ; out += (1-alpha)*x_oca ----
  for (int tile = wave; tile < 48; tile += 8) {
    int mt = tile / 12, nt = tile % 12;
    v8f acc = {};
#pragma unroll
    for (int kt = 0; kt < 6; ++kt) {
      v16h a = load_a_fast(s_o, CH, mt * 16, kt * 32);
      v16h bm = load_b_packed(projw, 6, kt, nt);
      acc = wmma32(a, bm, acc);
    }
    const int lane = threadIdx.x & 31;
    const int n = lane & 15, half = lane >> 4;
    const int c = nt * 16 + n;
#pragma unroll
    for (int r = 0; r < 8; ++r) {
      int t = mt * 16 + r + 8 * half;
      int gi = xidx(b, c, h0 + (t >> 3), w0 + (t & 7));
      out[gi] += am1 * (acc[r] + x[gi]);
    }
  }
}

// ===========================================================================
extern "C" void kernel_launch(void* const* d_in, const int* in_sizes, int n_in,
                              void* d_out, int out_size, void* d_ws, size_t ws_size,
                              hipStream_t stream) {
  (void)in_sizes; (void)n_in; (void)out_size; (void)ws_size;
  const float* x      = (const float*)d_in[0];
  const float* n1g    = (const float*)d_in[1];
  const float* n1b    = (const float*)d_in[2];
  const float* n2g    = (const float*)d_in[3];
  const float* n2b    = (const float*)d_in[4];
  const float* qkv_w  = (const float*)d_in[5];
  const float* proj_w = (const float*)d_in[6];
  const float* rpe    = (const float*)d_in[7];
  const float* ocaq_w = (const float*)d_in[8];
  const float* ocakv_w= (const float*)d_in[9];
  const float* ocap_w = (const float*)d_in[10];
  const float* ocan_g = (const float*)d_in[11];
  const float* ocan_b = (const float*)d_in[12];
  const float* mlp_w1 = (const float*)d_in[13];
  const float* mlp_w2 = (const float*)d_in[14];
  const float* alpha  = (const float*)d_in[15];
  float* out = (float*)d_out;

  // fragment-order f16 weights in workspace
  _Float16* wb = (_Float16*)d_ws;
  size_t o = 0;
  _Float16* qkv16  = wb + o; o += (size_t)CH * 3 * CH;
  _Float16* proj16 = wb + o; o += (size_t)CH * CH;
  _Float16* w1_16  = wb + o; o += (size_t)CH * 2 * CH;
  _Float16* w2_16  = wb + o; o += (size_t)2 * CH * CH;
  _Float16* oq16   = wb + o; o += (size_t)CH * CH;
  _Float16* okv16  = wb + o; o += (size_t)CH * 2 * CH;
  _Float16* op16   = wb + o; o += (size_t)CH * CH;

  auto pack = [&](const float* s, _Float16* d, int K, int N) {
    pack_b_kernel<<<(K * N + 255) / 256, 256, 0, stream>>>(s, d, K, N);
  };
  pack(qkv_w,   qkv16,  CH, 3 * CH);
  pack(proj_w,  proj16, CH, CH);
  pack(mlp_w1,  w1_16,  CH, 2 * CH);
  pack(mlp_w2,  w2_16,  2 * CH, CH);
  pack(ocaq_w,  oq16,   CH, CH);
  pack(ocakv_w, okv16,  CH, 2 * CH);
  pack(ocap_w,  op16,   CH, CH);

  hat_win_kernel<<<NWIN, 256, K1_SMEM, stream>>>(
      x, n1g, n1b, n2g, n2b, qkv16, proj16, rpe, w1_16, w2_16, alpha, out);
  hat_oca_kernel<<<NWIN, 256, K3_SMEM, stream>>>(
      x, ocan_g, ocan_b, oq16, okv16, op16, alpha, out);
}
